// CrossAttentionModule_53944789238506
// MI455X (gfx1250) — compile-verified, matrix operand order checked by reading, not run on hardware
//
#include <hip/hip_runtime.h>
#include <math.h>

typedef __bf16 bf16_t;
typedef __attribute__((ext_vector_type(16))) __bf16 v16bf;
typedef __attribute__((ext_vector_type(8)))  float  v8f;

#define DIM 128
#define HEADS 8
#define HEAD_DIM 16
#define RBF_K 16
#define ATTN_SCALE 0.25f   // HEAD_DIM^-0.5

#if __has_builtin(__builtin_amdgcn_global_load_async_to_lds_b128)
#define HAVE_ASYNC_LDS 1
#else
#define HAVE_ASYNC_LDS 0
#endif

#if HAVE_ASYNC_LDS
typedef int v4i __attribute__((vector_size(16)));
typedef __attribute__((address_space(1))) v4i* g_v4i_ptr;
typedef __attribute__((address_space(3))) v4i* l_v4i_ptr;
#endif

// ---------------------------------------------------------------------------
// float atomic max via signed-max / unsigned-min trick (init to -inf bits)
// ---------------------------------------------------------------------------
__device__ __forceinline__ void atomic_max_f32(float* addr, float val) {
    if (val >= 0.0f) atomicMax((int*)addr, __float_as_int(val));
    else             atomicMin((unsigned int*)addr, __float_as_uint(val));
}

__global__ void fill_f32(float* __restrict__ p, float val, int n) {
    int i = blockIdx.x * blockDim.x + threadIdx.x;
    if (i < n) p[i] = val;
}

// ---------------------------------------------------------------------------
// Pre-swizzle a 128x128 f32 weight into bf16 WMMA B-fragment layout.
// Fragment (kk, ntile): lane l holds 16 contiguous bf16 = its B column slice.
// Offset of lane data: ((kk*8 + ntile)*32 + lane)*16.  Grid: 4 x 256 threads.
// ---------------------------------------------------------------------------
__global__ __launch_bounds__(256) void swizzle_w(const float* __restrict__ W,
                                                 bf16_t* __restrict__ Wb) {
    int t = blockIdx.x * 256 + threadIdx.x;    // 0..1023 = (kk, ntile, lane)
    int lane  = t & 31;
    int ntile = (t >> 5) & 7;
    int kk    = t >> 8;
    int half  = lane >> 4;
    int col   = ntile * 16 + (lane & 15);
    v16bf tmp;
#pragma unroll
    for (int i = 0; i < 8; ++i) {
        int k0 = kk * 32 + (i & 3) * 2 + (i >> 2) * 16 + half * 8;
        tmp[2 * i]     = (bf16_t)W[(size_t)k0 * DIM + col];
        tmp[2 * i + 1] = (bf16_t)W[(size_t)(k0 + 1) * DIM + col];
    }
    *(v16bf*)(Wb + (size_t)t * 16) = tmp;
}

// ---------------------------------------------------------------------------
// Stage a 64x128 f32 tile into LDS (async DMA path when available)
// ---------------------------------------------------------------------------
__device__ __forceinline__ void stage_tile_64x128(float* xs, const float* __restrict__ X,
                                                  int row0, int M, int tid) {
#if HAVE_ASYNC_LDS
#pragma unroll
    for (int s = 0; s < 8; ++s) {
        int c   = s * 256 + tid;          // 16-byte chunk id, 2048 total
        int row = row0 + (c >> 5);
        int cc  = (c & 31) * 4;
        if (row >= M) row = M - 1;
        float* g = const_cast<float*>(X) + (size_t)row * DIM + cc;
        __builtin_amdgcn_global_load_async_to_lds_b128((g_v4i_ptr)g,
                                                       (l_v4i_ptr)(xs + (c << 2)),
                                                       0, 0);
    }
    asm volatile("s_wait_asynccnt 0x0" ::: "memory");
#else
#pragma unroll
    for (int s = 0; s < 32; ++s) {
        int lin = s * 256 + tid;
        int row = row0 + (lin >> 7);
        if (row >= M) row = M - 1;
        xs[lin] = X[(size_t)row * DIM + (lin & 127)];
    }
#endif
}

// ---------------------------------------------------------------------------
// Y[M,128] (bf16) = X[M,128](f32) @ W + b.  W pre-swizzled bf16 fragments.
// 64 rows per block; 8 waves own one 16-col tile each; B frags (4x v16bf)
// held in registers and reused across 4 M-tiles; 16 WMMAs per wave.
// ---------------------------------------------------------------------------
__global__ __launch_bounds__(256) void proj_wmma(const float* __restrict__ X,
                                                 const bf16_t* __restrict__ Wb,
                                                 const float* __restrict__ bias,
                                                 bf16_t* __restrict__ Y, int M) {
    __shared__ float xs[64 * DIM];       // 32 KB
    const int tid  = threadIdx.x;
    const int lane = tid & 31;
    const int wave = tid >> 5;           // N-tile
    const int half = lane >> 4;
    const int r    = lane & 15;
    const int row0 = blockIdx.x * 64;
    const int col  = wave * 16 + r;

    stage_tile_64x128(xs, X, row0, M, tid);
    __syncthreads();

    v16bf bw[4];
#pragma unroll
    for (int kk = 0; kk < 4; ++kk)
        bw[kk] = *(const v16bf*)(Wb + (((size_t)kk * 8 + wave) * 32 + lane) * 16);

    const float bc = bias[col];
#pragma unroll
    for (int m = 0; m < 4; ++m) {
        v8f acc = {};
#pragma unroll
        for (int kk = 0; kk < 4; ++kk) {
            v16bf a;
#pragma unroll
            for (int i = 0; i < 8; ++i) {
                int k0 = kk * 32 + (i & 3) * 2 + (i >> 2) * 16 + half * 8;
                a[2 * i]     = (bf16_t)xs[(m * 16 + r) * DIM + k0];
                a[2 * i + 1] = (bf16_t)xs[(m * 16 + r) * DIM + k0 + 1];
            }
            acc = __builtin_amdgcn_wmma_f32_16x16x32_bf16(false, a, false, bw[kk],
                                                          (short)0, acc, false, false);
        }
#pragma unroll
        for (int v = 0; v < 8; ++v) {
            int mrow = row0 + m * 16 + v + half * 8;
            if (mrow < M) Y[(size_t)mrow * DIM + col] = (bf16_t)(acc[v] + bc);
        }
    }
}

// ---------------------------------------------------------------------------
// Out[M,128](f32) = (NUM * inv(S))[M,128] @ Wo + bo, normalization folded in.
// ---------------------------------------------------------------------------
__global__ __launch_bounds__(256) void outproj_wmma(const float* __restrict__ NUM,
                                                    const float* __restrict__ S,
                                                    const bf16_t* __restrict__ Wb,
                                                    const float* __restrict__ bias,
                                                    float* __restrict__ Y, int M) {
    __shared__ float xs[64 * DIM];       // 32 KB
    __shared__ float inv[64 * HEADS];    // 2 KB
    const int tid  = threadIdx.x;
    const int lane = tid & 31;
    const int wave = tid >> 5;
    const int half = lane >> 4;
    const int r    = lane & 15;
    const int row0 = blockIdx.x * 64;
    const int col  = wave * 16 + r;

    stage_tile_64x128(xs, NUM, row0, M, tid);
#pragma unroll
    for (int s = 0; s < 2; ++s) {
        int idx = s * 256 + tid;         // 512 (row, head) slots
        int row = row0 + (idx >> 3);
        float sv = (row < M) ? S[(size_t)row * HEADS + (idx & 7)] : 0.0f;
        inv[idx] = 1.0f / (sv + 1e-8f);
    }
    __syncthreads();

    v16bf bw[4];
#pragma unroll
    for (int kk = 0; kk < 4; ++kk)
        bw[kk] = *(const v16bf*)(Wb + (((size_t)kk * 8 + wave) * 32 + lane) * 16);

    const float bc = bias[col];
#pragma unroll
    for (int m = 0; m < 4; ++m) {
        v8f acc = {};
#pragma unroll
        for (int kk = 0; kk < 4; ++kk) {
            v16bf a;
#pragma unroll
            for (int i = 0; i < 8; ++i) {
                int k0 = kk * 32 + (i & 3) * 2 + (i >> 2) * 16 + half * 8;
                float iv = inv[(m * 16 + r) * HEADS + (k0 >> 4)];  // k0,k0+1 same head
                a[2 * i]     = (bf16_t)(xs[(m * 16 + r) * DIM + k0] * iv);
                a[2 * i + 1] = (bf16_t)(xs[(m * 16 + r) * DIM + k0 + 1] * iv);
            }
            acc = __builtin_amdgcn_wmma_f32_16x16x32_bf16(false, a, false, bw[kk],
                                                          (short)0, acc, false, false);
        }
#pragma unroll
        for (int v = 0; v < 8; ++v) {
            int mrow = row0 + m * 16 + v + half * 8;
            if (mrow < M) Y[(size_t)mrow * DIM + col] = acc[v] + bc;
        }
    }
}

// ---------------------------------------------------------------------------
// Per (edge, head): both direction logits (incl. rbf bias) + atomic seg-max
// ---------------------------------------------------------------------------
__global__ void edge_logits(const int* __restrict__ edges,
                            const float* __restrict__ attr,
                            const float* __restrict__ Wr,
                            const float* __restrict__ br,
                            const bf16_t* __restrict__ ql,
                            const bf16_t* __restrict__ kp,
                            const bf16_t* __restrict__ qp,
                            const bf16_t* __restrict__ kl,
                            const int* __restrict__ lig_start,
                            const int* __restrict__ prot_start,
                            float* __restrict__ logits_l,
                            float* __restrict__ logits_p,
                            float* __restrict__ m_l,
                            float* __restrict__ m_p, int E) {
    int t = blockIdx.x * blockDim.x + threadIdx.x;
    if (t >= E * HEADS) return;
    int e = t >> 3, h = t & 7;
    int li = edges[2 * e]     - lig_start[0];
    int pi = edges[2 * e + 1] - prot_start[0];

    float bias = br[h];
#pragma unroll
    for (int i = 0; i < RBF_K; ++i)
        bias += attr[(size_t)e * RBF_K + i] * Wr[i * HEADS + h];

    v16bf qv1 = *(const v16bf*)(ql + (size_t)li * DIM + h * HEAD_DIM);
    v16bf kv1 = *(const v16bf*)(kp + (size_t)pi * DIM + h * HEAD_DIM);
    v16bf qv2 = *(const v16bf*)(qp + (size_t)pi * DIM + h * HEAD_DIM);
    v16bf kv2 = *(const v16bf*)(kl + (size_t)li * DIM + h * HEAD_DIM);
    float d1 = 0.f, d2 = 0.f;
#pragma unroll
    for (int d = 0; d < HEAD_DIM; ++d) {
        d1 += (float)qv1[d] * (float)kv1[d];
        d2 += (float)qv2[d] * (float)kv2[d];
    }
    float lg1 = d1 * ATTN_SCALE + bias;
    float lg2 = d2 * ATTN_SCALE + bias;
    logits_l[t] = lg1;
    logits_p[t] = lg2;
    atomic_max_f32(m_l + (size_t)li * HEADS + h, lg1);
    atomic_max_f32(m_p + (size_t)pi * HEADS + h, lg2);
}

// ---------------------------------------------------------------------------
// Per (edge, head): ex = exp(logit - m[dst]); s[dst] += ex; num[dst] += ex*v[src]
// ---------------------------------------------------------------------------
__global__ void edge_accum(const int* __restrict__ edges,
                           const float* __restrict__ logits,
                           const float* __restrict__ m,
                           float* __restrict__ s,
                           float* __restrict__ num,
                           const bf16_t* __restrict__ v,
                           const int* __restrict__ dst_start,
                           const int* __restrict__ src_start,
                           int dst_col, int src_col, int E) {
    int t = blockIdx.x * blockDim.x + threadIdx.x;
    if (t >= E * HEADS) return;
    int e = t >> 3, h = t & 7;
    int dst = edges[2 * e + dst_col] - dst_start[0];
    int src = edges[2 * e + src_col] - src_start[0];
    float ex = __expf(logits[t] - m[(size_t)dst * HEADS + h]);
    atomicAdd(&s[(size_t)dst * HEADS + h], ex);
    v16bf vv = *(const v16bf*)(v + (size_t)src * DIM + h * HEAD_DIM);
    float* np = num + (size_t)dst * DIM + h * HEAD_DIM;
#pragma unroll
    for (int d = 0; d < HEAD_DIM; ++d)
        atomicAdd(np + d, ex * (float)vv[d]);
}

// ---------------------------------------------------------------------------
extern "C" void kernel_launch(void* const* d_in, const int* in_sizes, int n_in,
                              void* d_out, int out_size, void* d_ws, size_t ws_size,
                              hipStream_t stream) {
    const float* ligand_x   = (const float*)d_in[0];
    const float* protein_x  = (const float*)d_in[1];
    const int*   edges      = (const int*)d_in[2];
    const float* attr       = (const float*)d_in[3];
    const float* Wq = (const float*)d_in[4];  const float* bq = (const float*)d_in[5];
    const float* Wk = (const float*)d_in[6];  const float* bk = (const float*)d_in[7];
    const float* Wv = (const float*)d_in[8];  const float* bv = (const float*)d_in[9];
    const float* Wo = (const float*)d_in[10]; const float* bo = (const float*)d_in[11];
    const float* Wr = (const float*)d_in[12]; const float* br = (const float*)d_in[13];
    const int* lig_start  = (const int*)d_in[14];
    const int* prot_start = (const int*)d_in[15];

    const int n_lig  = in_sizes[0] / DIM;
    const int n_prot = in_sizes[1] / DIM;
    const int E      = in_sizes[2] / 2;

    // ---- workspace carve ----
    char*  ws  = (char*)d_ws;
    size_t off = 0;
    auto carve = [&](size_t bytes) -> void* {
        void* p = ws + off;
        off = (off + bytes + 255) & ~(size_t)255;
        return p;
    };
    bf16_t* q_l = (bf16_t*)carve((size_t)n_lig  * DIM * sizeof(bf16_t));
    bf16_t* k_l = (bf16_t*)carve((size_t)n_lig  * DIM * sizeof(bf16_t));
    bf16_t* v_l = (bf16_t*)carve((size_t)n_lig  * DIM * sizeof(bf16_t));
    bf16_t* q_p = (bf16_t*)carve((size_t)n_prot * DIM * sizeof(bf16_t));
    bf16_t* k_p = (bf16_t*)carve((size_t)n_prot * DIM * sizeof(bf16_t));
    bf16_t* v_p = (bf16_t*)carve((size_t)n_prot * DIM * sizeof(bf16_t));
    float* logits_l = (float*)carve((size_t)E * HEADS * sizeof(float));
    float* logits_p = (float*)carve((size_t)E * HEADS * sizeof(float));
    float* m_l = (float*)carve((size_t)n_lig  * HEADS * sizeof(float));
    float* s_l = (float*)carve((size_t)n_lig  * HEADS * sizeof(float));
    float* m_p = (float*)carve((size_t)n_prot * HEADS * sizeof(float));
    float* s_p = (float*)carve((size_t)n_prot * HEADS * sizeof(float));
    float* num_l = (float*)carve((size_t)n_lig  * DIM * sizeof(float));
    float* num_p = (float*)carve((size_t)n_prot * DIM * sizeof(float));
    bf16_t* Wqb = (bf16_t*)carve(4 * 8 * 32 * 16 * sizeof(bf16_t));  // 32 KB each
    bf16_t* Wkb = (bf16_t*)carve(4 * 8 * 32 * 16 * sizeof(bf16_t));
    bf16_t* Wvb = (bf16_t*)carve(4 * 8 * 32 * 16 * sizeof(bf16_t));
    bf16_t* Wob = (bf16_t*)carve(4 * 8 * 32 * 16 * sizeof(bf16_t));

    const float NEG_INF = -__builtin_huge_valf();
    auto nblk = [](int n) { return (n + 255) / 256; };

    // ---- init accumulators (fresh every call) ----
    fill_f32<<<nblk(n_lig  * HEADS), 256, 0, stream>>>(m_l, NEG_INF, n_lig  * HEADS);
    fill_f32<<<nblk(n_prot * HEADS), 256, 0, stream>>>(m_p, NEG_INF, n_prot * HEADS);
    fill_f32<<<nblk(n_lig  * HEADS), 256, 0, stream>>>(s_l, 0.0f, n_lig  * HEADS);
    fill_f32<<<nblk(n_prot * HEADS), 256, 0, stream>>>(s_p, 0.0f, n_prot * HEADS);
    fill_f32<<<nblk(n_lig  * DIM),   256, 0, stream>>>(num_l, 0.0f, n_lig  * DIM);
    fill_f32<<<nblk(n_prot * DIM),   256, 0, stream>>>(num_p, 0.0f, n_prot * DIM);

    // ---- weight swizzle (f32 -> bf16 WMMA B-fragment layout) ----
    swizzle_w<<<4, 256, 0, stream>>>(Wq, Wqb);
    swizzle_w<<<4, 256, 0, stream>>>(Wk, Wkb);
    swizzle_w<<<4, 256, 0, stream>>>(Wv, Wvb);
    swizzle_w<<<4, 256, 0, stream>>>(Wo, Wob);

    // ---- projections (WMMA, 64 rows/block) ----
    const int mt_l = (n_lig  + 63) / 64;
    const int mt_p = (n_prot + 63) / 64;
    proj_wmma<<<mt_l, 256, 0, stream>>>(ligand_x,  Wqb, bq, q_l, n_lig);
    proj_wmma<<<mt_l, 256, 0, stream>>>(ligand_x,  Wkb, bk, k_l, n_lig);
    proj_wmma<<<mt_l, 256, 0, stream>>>(ligand_x,  Wvb, bv, v_l, n_lig);
    proj_wmma<<<mt_p, 256, 0, stream>>>(protein_x, Wqb, bq, q_p, n_prot);
    proj_wmma<<<mt_p, 256, 0, stream>>>(protein_x, Wkb, bk, k_p, n_prot);
    proj_wmma<<<mt_p, 256, 0, stream>>>(protein_x, Wvb, bv, v_p, n_prot);

    // ---- edge logits + segment max ----
    edge_logits<<<nblk(E * HEADS), 256, 0, stream>>>(
        edges, attr, Wr, br, q_l, k_p, q_p, k_l,
        lig_start, prot_start, logits_l, logits_p, m_l, m_p, E);

    // ---- exp + segment sums (both directions) ----
    edge_accum<<<nblk(E * HEADS), 256, 0, stream>>>(
        edges, logits_l, m_l, s_l, num_l, v_p, lig_start, prot_start, 0, 1, E);
    edge_accum<<<nblk(E * HEADS), 256, 0, stream>>>(
        edges, logits_p, m_p, s_p, num_p, v_l, prot_start, lig_start, 1, 0, E);

    // ---- normalize + output projection (WMMA), straight into d_out ----
    float* out = (float*)d_out;
    outproj_wmma<<<mt_l, 256, 0, stream>>>(num_l, s_l, Wob, bo, out, n_lig);
    outproj_wmma<<<mt_p, 256, 0, stream>>>(num_p, s_p, Wob, bo,
                                           out + (size_t)n_lig * DIM, n_prot);
}
